// MultiTurnCRF_24584392802475
// MI455X (gfx1250) — matrix operand order repaired and checked by reading, not compiled, not run on hardware
//
#include <hip/hip_runtime.h>

// ---------------------------------------------------------------------------
// MultiTurnCRF on MI455X (gfx1250).
//
// log Z: forward recursion in exp-domain is a linear recurrence
//   a_t = a_{t-1} @ G_t,  G_t[i][j] = exp(tr[idx_{t-1}][i][j] + em_t[j])
// Chunked parallel scan:
//   Phase 1: C chunks in parallel; each block builds its chunk product
//            transposed: P'^T = E^T @ P^T via 128x128x128 bf16 WMMA
//            (v_wmma_f32_16x16x32_bf16). E^T = exp(tr-max) for the 4
//            transition matrices is LDS-resident (bf16, 128 KB) -- needs
//            CDNA5's 320 KB/WGP LDS. The running product lives ONLY as the
//            bf16 WMMA operand; renormalization max is computed from the
//            accumulator registers and folded into the packed b128 stores,
//            so the step loop is: 1 global em load + 2 shuffle reductions +
//            72 vector ds loads + 32 WMMA + 8 vector ds stores per wave.
//   Phase 2: log2(C) tree levels of WMMA matmuls combine chunk products.
//   Phase 3: log Z = a0max + s_total + log(sum_i exp(a0_i-a0max)*rowsum_i).
// Gold score: parallel gather + deterministic two-stage sum.
// Total matrix work = T * 2K^3 ~ 5.5e11 FLOP -> O(100us) at bf16-WMMA rates;
// any scalar sequential scan is >10ms. Emissions (67MB) stream once (~3us
// at 23.3 TB/s).
// ---------------------------------------------------------------------------

#define T_LEN   131072
#define KDIM    128
#define NSTEPS  (T_LEN - 1)   // 131071
#define TPB     256
#define GOLD_BLOCKS 512

typedef __attribute__((ext_vector_type(16))) __bf16 v16bf;
typedef __attribute__((ext_vector_type(8)))  __bf16 v8bf;
typedef __attribute__((ext_vector_type(8)))  float  v8f;

// LDS layout for the chunk kernel (bytes)
#define SH_E   (4 * KDIM * KDIM * 2)          // 131072 : exp(tr)^T bf16  [mat][j][i]
#define SH_PB  (KDIM * KDIM * 2)              //  32768 : P^T operand, bf16 row-major [row][col]
#define SH_EMX (KDIM * 4)                     //    512 : exp(em - max)
#define SH_RED 64                             //     64 : 8-slot reduction scratch (padded)
#define SH_MXV 16                             //     16 : per-matrix max
#define SHM_CHUNK (SH_E + SH_PB + SH_EMX + SH_RED + SH_MXV)   // 164432
#define SHM_TREE  (KDIM*KDIM*2 + KDIM*KDIM*2 + SH_RED)        //  65600

// ---- bf16 helpers (explicit RNE, no reliance on __bf16 arithmetic) --------
static __device__ __forceinline__ __bf16 f2bf(float f) {
    unsigned u = __float_as_uint(f);
    unsigned r = (u + 0x7FFFu + ((u >> 16) & 1u)) >> 16;
    return __builtin_bit_cast(__bf16, (unsigned short)r);
}
static __device__ __forceinline__ float bf2f(__bf16 h) {
    unsigned short s = __builtin_bit_cast(unsigned short, h);
    return __uint_as_float(((unsigned)s) << 16);
}

// ---- reductions: shuffle within wave32, 8-slot LDS across 8 waves ---------
static __device__ __forceinline__ float wave_max(float v) {
#pragma unroll
    for (int m = 16; m > 0; m >>= 1) v = fmaxf(v, __shfl_xor(v, m, 32));
    return v;
}
static __device__ __forceinline__ float wave_sum(float v) {
#pragma unroll
    for (int m = 16; m > 0; m >>= 1) v += __shfl_xor(v, m, 32);
    return v;
}
static __device__ __forceinline__ float blk_max8(float v, float* red8) {
    v = wave_max(v);
    if ((threadIdx.x & 31) == 0) red8[threadIdx.x >> 5] = v;
    __syncthreads();
    float r = red8[0];
#pragma unroll
    for (int i = 1; i < 8; ++i) r = fmaxf(r, red8[i]);
    __syncthreads();
    return r;
}
static __device__ __forceinline__ float blk_sum8(float v, float* red8) {
    v = wave_sum(v);
    if ((threadIdx.x & 31) == 0) red8[threadIdx.x >> 5] = v;
    __syncthreads();
    float r = 0.0f;
#pragma unroll
    for (int i = 0; i < 8; ++i) r += red8[i];
    __syncthreads();
    return r;
}

static __device__ __forceinline__ int trans_idx(const int* __restrict__ w2w,
                                                const int* __restrict__ ic,
                                                const int* __restrict__ dist, int u) {
    int w = w2w[u];
    int i = ic[u];
    int d = dist[u];
    return (w == 1) ? 0 : (i == 0 ? 1 : (d == 0 ? 2 : 3));
}

// ---- 128x128x128 block matmul on WMMA -------------------------------------
// Computes D = A @ (Bt)^T where both operands are stored with the K index
// contiguous:  A[m*128+k],  Bt[n*128+k].  8 waves; wave w owns output tile
// rows [16w,16w+16); acc[nt] is one 16x16 f32 tile (N-tile nt).
static __device__ __forceinline__ void mm128_wmma(const __bf16* __restrict__ A,
                                                  const __bf16* __restrict__ Bt,
                                                  v8f acc[8]) {
    int lane = threadIdx.x & 31;
    int w    = threadIdx.x >> 5;
    int row  = (w << 4) + (lane & 15);
    int offA = (lane & 16) ? 16 : 0;   // A-frag K-offset (contiguous k = 16h + s, matches B)
    int offB = (lane & 16) ? 16 : 0;   // B-frag K-offset
    int n0   = lane & 15;

    v16bf afr[4];
#pragma unroll
    for (int kk = 0; kk < 4; ++kk) {
        const __bf16* pa = A + row * KDIM + kk * 32 + offA;
        v8bf lo = *(const v8bf*)(pa);
        v8bf hi = *(const v8bf*)(pa + 8);
        v16bf a;
#pragma unroll
        for (int e = 0; e < 8; ++e) { a[e] = lo[e]; a[e + 8] = hi[e]; }
        afr[kk] = a;
    }
#pragma unroll
    for (int nt = 0; nt < 8; ++nt) {
        const __bf16* pb = Bt + (nt * 16 + n0) * KDIM + offB;
        v8f c = acc[nt];
#pragma unroll
        for (int kk = 0; kk < 4; ++kk) {
            v16bf b = *(const v16bf*)(pb + kk * 32);
            c = __builtin_amdgcn_wmma_f32_16x16x32_bf16(
                    false, afr[kk], false, b, (short)0, c, false, false);
        }
        acc[nt] = c;
    }
}

// ===========================================================================
// Phase 1: per-chunk product of step matrices (one block per chunk).
// The running product is maintained TRANSPOSED in bf16 (Pbf[row][col] holds
// P[row][col], used as the B operand of  P'^T = E^T @ P^T ).
// ===========================================================================
__global__ __launch_bounds__(TPB)
void crf_chunk_kernel(const float* __restrict__ em,
                      const float* __restrict__ trans,
                      const int* __restrict__ w2w,
                      const int* __restrict__ ic,
                      const int* __restrict__ dist,
                      int L,
                      float* __restrict__ matsOut,
                      float* __restrict__ sOut) {
    extern __shared__ char smem[];
    __bf16* EbfT = (__bf16*)(smem);                    // [4][j][i] = E^T row-major
    __bf16* Pbf  = (__bf16*)(smem + SH_E);             // [128][128] = P row-major
    float*  emx  = (float*)(smem + SH_E + SH_PB);      // [128]
    float*  red8 = emx + KDIM;                         // [8] (+pad)
    float*  mxv  = red8 + 16;                          // [4]

    const int tid = threadIdx.x;
    const int c   = blockIdx.x;

    int t0   = c * L + 1;
    int tend = (c + 1) * L; if (tend > NSTEPS) tend = NSTEPS;

    if (t0 > NSTEPS) {  // degenerate chunk -> identity
        for (int idx = tid; idx < KDIM * KDIM; idx += TPB) {
            int i = idx >> 7, j = idx & 127;
            matsOut[(size_t)c * KDIM * KDIM + idx] = (i == j) ? 1.0f : 0.0f;
        }
        if (tid == 0) sOut[c] = 0.0f;
        return;
    }

    // per-matrix max of trans_stack, then exp(tr - max) -> LDS transposed bf16
    for (int mat = 0; mat < 4; ++mat) {
        float lm = -3.0e38f;
        for (int i = tid; i < KDIM * KDIM; i += TPB)
            lm = fmaxf(lm, trans[mat * KDIM * KDIM + i]);
        float M = blk_max8(lm, red8);
        if (tid == 0) mxv[mat] = M;
    }
    __syncthreads();
    for (int idx = tid; idx < 4 * KDIM * KDIM; idx += TPB) {
        int mat = idx >> 14, rem = idx & (KDIM * KDIM - 1);
        int i = rem >> 7, j = rem & 127;
        EbfT[mat * KDIM * KDIM + j * KDIM + i] = f2bf(__expf(trans[idx] - mxv[mat]));
    }
    __syncthreads();

    float s = 0.0f;

    // first step: P = G_{t0} built elementwise (entries <= 1, no renorm needed)
    {
        int sel = trans_idx(w2w, ic, dist, t0 - 1);
        float v = (tid < KDIM) ? em[(size_t)t0 * KDIM + tid] : -3.0e38f;
        float m = blk_max8(v, red8);
        if (tid < KDIM) emx[tid] = __expf(v - m);
        s += m + mxv[sel];
        __syncthreads();
        const __bf16* E = EbfT + sel * KDIM * KDIM;
        // iterate in EbfT order (contiguous reads), scattered Pbf writes
        for (int idx = tid; idx < KDIM * KDIM; idx += TPB) {
            int j = idx >> 7, i = idx & 127;       // EbfT[j][i] = E[i][j]
            Pbf[i * KDIM + j] = f2bf(bf2f(E[idx]) * emx[j]);
        }
    }

    const int lane = tid & 31, w = tid >> 5;
    const int hi8 = (lane & 16) ? 8 : 0, nl = lane & 15;
    const int cb  = (w << 4) + hi8;   // 8 consecutive output columns cb..cb+7

    for (int t = t0 + 1; t <= tend; ++t) {
        // emission row for this step (the reduction's barrier also fences
        // the previous step's Pbf stores before this step's matmul reads)
        int sel = trans_idx(w2w, ic, dist, t - 1);
        float v = (tid < KDIM) ? em[(size_t)t * KDIM + tid] : -3.0e38f;
        if (t < tend)
            __builtin_prefetch(em + (size_t)(t + 1) * KDIM + (tid & 127), 0, 3);
        float mt = blk_max8(v, red8);
        if (tid < KDIM) emx[tid] = __expf(v - mt);
        s += mt + mxv[sel];

        // P'^T = E_sel^T @ P^T
        v8f acc[8];
#pragma unroll
        for (int nt = 0; nt < 8; ++nt)
#pragma unroll
            for (int r = 0; r < 8; ++r) acc[nt][r] = 0.0f;

        mm128_wmma(EbfT + sel * KDIM * KDIM, Pbf, acc);

        // renorm max straight from the accumulators (emx <= 1 so this bounds
        // the scaled values); its barrier also fences Pbf reads vs stores
        float lm = 0.0f;
#pragma unroll
        for (int nt = 0; nt < 8; ++nt)
#pragma unroll
            for (int r = 0; r < 8; ++r) lm = fmaxf(lm, acc[nt][r]);
        float Mx = fmaxf(blk_max8(lm, red8), 1e-30f);
        float inv = 1.0f / Mx;
        s += logf(Mx);

        // packed b128 stores: D tile rows (r), i.e. P' columns, are contiguous in Pbf rows
        v8f ev = *(const v8f*)(emx + cb);
#pragma unroll
        for (int nt = 0; nt < 8; ++nt) {
            int rho = (nt << 4) + nl;             // output row (of P')
            v8bf pk;
#pragma unroll
            for (int r = 0; r < 8; ++r) pk[r] = f2bf(acc[nt][r] * ev[r] * inv);
            *(v8bf*)(Pbf + rho * KDIM + cb) = pk;
        }
    }
    __syncthreads();

    for (int idx = tid; idx < KDIM * KDIM; idx += TPB)
        matsOut[(size_t)c * KDIM * KDIM + idx] = bf2f(Pbf[idx]);
    if (tid == 0) sOut[c] = s;
}

// ===========================================================================
// Phase 2: pairwise tree combine of chunk products (one block per pair)
// ===========================================================================
__global__ __launch_bounds__(TPB)
void crf_tree_kernel(const float* __restrict__ srcM,
                     const float* __restrict__ srcS,
                     float* __restrict__ dstM,
                     float* __restrict__ dstS) {
    extern __shared__ char smem[];
    __bf16* Abf  = (__bf16*)(smem);                       // row-major [m][k]
    __bf16* BbfT = (__bf16*)(smem + KDIM * KDIM * 2);     // transposed [n][k]
    float*  red8 = (float*)(smem + 2 * KDIM * KDIM * 2);

    const int tid = threadIdx.x;
    const int b   = blockIdx.x;
    const float* Ag = srcM + (size_t)(2 * b)     * KDIM * KDIM;
    const float* Bg = srcM + (size_t)(2 * b + 1) * KDIM * KDIM;

    float lmA = 0.0f, lmB = 0.0f;
    for (int i = tid; i < KDIM * KDIM; i += TPB) lmA = fmaxf(lmA, Ag[i]);
    float MA = fmaxf(blk_max8(lmA, red8), 1e-30f);
    for (int i = tid; i < KDIM * KDIM; i += TPB) lmB = fmaxf(lmB, Bg[i]);
    float MB = fmaxf(blk_max8(lmB, red8), 1e-30f);
    float invA = 1.0f / MA, invB = 1.0f / MB;
    for (int i = tid; i < KDIM * KDIM; i += TPB)
        Abf[i] = f2bf(Ag[i] * invA);
    for (int i = tid; i < KDIM * KDIM; i += TPB) {
        int r = i >> 7, cc = i & 127;
        BbfT[cc * KDIM + r] = f2bf(Bg[i] * invB);
    }
    __syncthreads();

    v8f acc[8];
#pragma unroll
    for (int nt = 0; nt < 8; ++nt)
#pragma unroll
        for (int r = 0; r < 8; ++r) acc[nt][r] = 0.0f;

    mm128_wmma(Abf, BbfT, acc);

    float lm = 0.0f;
#pragma unroll
    for (int nt = 0; nt < 8; ++nt)
#pragma unroll
        for (int r = 0; r < 8; ++r) lm = fmaxf(lm, acc[nt][r]);
    float MC = fmaxf(blk_max8(lm, red8), 1e-30f);
    float invC = 1.0f / MC;

    int lane = tid & 31, w = tid >> 5;
    int hi8 = (lane & 16) ? 8 : 0, nl = lane & 15;
#pragma unroll
    for (int nt = 0; nt < 8; ++nt)
#pragma unroll
        for (int r = 0; r < 8; ++r) {
            int M = (w << 4) + r + hi8;
            int N = nt * 16 + nl;
            dstM[(size_t)b * KDIM * KDIM + M * KDIM + N] = acc[nt][r] * invC;
        }
    if (tid == 0)
        dstS[b] = srcS[2 * b] + srcS[2 * b + 1] + logf(MA) + logf(MB) + logf(MC);
}

// ===========================================================================
// Gold score partials (deterministic two-stage reduction)
// ===========================================================================
__global__ __launch_bounds__(TPB)
void crf_gold_kernel(const float* __restrict__ em,
                     const float* __restrict__ trans,
                     const float* __restrict__ start,
                     const float* __restrict__ cw,
                     const int* __restrict__ tags,
                     const int* __restrict__ w2w,
                     const int* __restrict__ ic,
                     const int* __restrict__ dist,
                     float* __restrict__ goldPart) {
    __shared__ float red8[16];
    const int tid = threadIdx.x;
    const int gid = blockIdx.x * TPB + tid;
    const int stride = gridDim.x * TPB;
    float local = 0.0f;
    for (int t = gid; t < T_LEN; t += stride) {
        int tg = tags[t];
        float wgt = cw[tg];
        float eg = em[(size_t)t * KDIM + tg];
        if (t == 0) {
            local += wgt * (start[tg] + eg);
        } else {
            int sel = trans_idx(w2w, ic, dist, t - 1);
            float tr = trans[sel * KDIM * KDIM + tags[t - 1] * KDIM + tg];
            local += wgt * (tr + eg);
        }
    }
    float tot = blk_sum8(local, red8);
    if (tid == 0) goldPart[blockIdx.x] = tot;
}

// ===========================================================================
// Final: gold sum + log Z = a0max + s_total + log(sum exp(a0-a0max)*rowsum)
// ===========================================================================
__global__ __launch_bounds__(TPB)
void crf_final_kernel(const float* __restrict__ Pfin,
                      const float* __restrict__ sFin,
                      const float* __restrict__ em,
                      const float* __restrict__ start,
                      const float* __restrict__ goldPart,
                      int ngold,
                      float* __restrict__ out) {
    __shared__ float red8[16];
    const int tid = threadIdx.x;
    if (tid == 0) {
        float g = 0.0f;
        for (int i = 0; i < ngold; ++i) g += goldPart[i];
        out[0] = g;
    }
    float a0 = (tid < KDIM) ? (start[tid] + em[tid]) : -3.0e38f;
    float am = blk_max8(a0, red8);
    float val = 0.0f;
    if (tid < KDIM) {
        float r = 0.0f;
        for (int j = 0; j < KDIM; ++j) r += Pfin[tid * KDIM + j];
        val = __expf(a0 - am) * r;
    }
    float sum = blk_sum8(val, red8);
    if (tid == 0) out[1] = am + sFin[0] + logf(fmaxf(sum, 1e-38f));
}

// ===========================================================================
extern "C" void kernel_launch(void* const* d_in, const int* in_sizes, int n_in,
                              void* d_out, int out_size, void* d_ws, size_t ws_size,
                              hipStream_t stream) {
    const float* emissions = (const float*)d_in[0];
    const float* trans     = (const float*)d_in[1];
    const float* start     = (const float*)d_in[2];
    const float* cw        = (const float*)d_in[3];
    const int*   tags      = (const int*)d_in[4];
    const int*   w2w       = (const int*)d_in[5];
    const int*   ic        = (const int*)d_in[6];
    const int*   dist      = (const int*)d_in[7];

    // pick C (power of two) to fit workspace: 2 ping-pong matrix regions +
    // 2 offset arrays + gold partials
    size_t floatsAvail = ws_size / sizeof(float);
    int C = 512;
    while (C > 8 &&
           (size_t)(2 * (size_t)C * KDIM * KDIM + 2 * (size_t)C + GOLD_BLOCKS) > floatsAvail)
        C >>= 1;
    int L = (NSTEPS + C - 1) / C;

    float* matsA    = (float*)d_ws;
    float* matsB    = matsA + (size_t)C * KDIM * KDIM;
    float* sA       = matsB + (size_t)C * KDIM * KDIM;
    float* sB       = sA + C;
    float* goldPart = sB + C;

    crf_gold_kernel<<<GOLD_BLOCKS, TPB, 0, stream>>>(
        emissions, trans, start, cw, tags, w2w, ic, dist, goldPart);

    crf_chunk_kernel<<<C, TPB, SHM_CHUNK, stream>>>(
        emissions, trans, w2w, ic, dist, L, matsA, sA);

    float* curM = matsA; float* curS = sA;
    float* othM = matsB; float* othS = sB;
    int n = C;
    while (n > 1) {
        int np = n >> 1;
        crf_tree_kernel<<<np, TPB, SHM_TREE, stream>>>(curM, curS, othM, othS);
        float* tM = curM; curM = othM; othM = tM;
        float* tS = curS; curS = othS; othS = tS;
        n = np;
    }

    crf_final_kernel<<<1, TPB, 0, stream>>>(
        curM, curS, emissions, start, goldPart, GOLD_BLOCKS, (float*)d_out);
}
